// MegrezMoeMoE_86320252715779
// MI455X (gfx1250) — matrix operand order repaired and checked by reading, not compiled, not run on hardware
//
#include <hip/hip_runtime.h>
#include <hip/hip_bf16.h>
#include <stdint.h>

// ---------------- problem constants ----------------
#define T_TOK 2048
#define H_DIM 2048
#define E_NUM 32
#define K_TOP 6
#define G_NUM 8
#define TOPKG 4
#define I_DIM 1408
#define IS_DIM 2816
#define SCALE_F 2.5f
#define LBS 40   // transposed B LDS stride (u16): 80B col pitch -> 16B aligned, bank-spread

typedef __attribute__((ext_vector_type(16))) __bf16 v16bf;
typedef __attribute__((ext_vector_type(8)))  float  v8f;

union ABFrag { v16bf v; uint16_t u[16]; uint4 q[2]; };
union Pack4 { uint16_t h[4]; unsigned long long q; };
union Pack8 { uint16_t h[8]; uint4 q; };

// fp32 -> bf16 round-to-nearest-even (bit level)
__device__ __forceinline__ uint16_t f2bf(float f) {
  union { float f; uint32_t u; } a; a.f = f;
  uint32_t r = a.u + 0x7FFFu + ((a.u >> 16) & 1u);
  return (uint16_t)(r >> 16);
}

// fast silu(g)*u : one v_exp + one v_rcp instead of an IEEE division ladder
__device__ __forceinline__ float silu_mul(float g, float u) {
  return g * __builtin_amdgcn_rcpf(1.f + __expf(-g)) * u;
}

// ---- CDNA5 async copy: global -> LDS, 16B per lane, ASYNCcnt-tracked ----
__device__ __forceinline__ void async_b128(uint32_t lds_off, const void* g) {
  asm volatile("global_load_async_to_lds_b128 %0, %1, off"
               :: "v"(lds_off), "v"((unsigned long long)(uintptr_t)g)
               : "memory");
}
__device__ __forceinline__ void wait_asynccnt0() {
  asm volatile("s_wait_asynccnt 0" ::: "memory");
}
__device__ __forceinline__ uint32_t lds_addr(const void* p) {
  return (uint32_t)(uintptr_t)p;   // low 32 bits of LDS-aperture generic ptr
}

// ---- WMMA fragment builders (ISA 7.12.2 layouts, 16-bit, wave32) ----
__device__ __forceinline__ void load_afrag(const uint16_t* lA, int mo, int lane, ABFrag& a) {
  const uint16_t* p = lA + (mo + (lane & 15)) * 32 + ((lane >> 4) << 3);
  a.q[0] = *(const uint4*)p;
  a.q[1] = *(const uint4*)(p + 16);
}
__device__ __forceinline__ void load_bfragT(const uint16_t* lBt, int no, int lane, ABFrag& b) {
  const uint16_t* p = lBt + (no + (lane & 15)) * LBS + ((lane >> 4) << 3);
  b.q[0] = *(const uint4*)p;
  b.q[1] = *(const uint4*)(p + 16);
}
__device__ __forceinline__ v8f wmma_bf16(const ABFrag& a, const ABFrag& b, v8f c) {
  return __builtin_amdgcn_wmma_f32_16x16x32_bf16(false, a.v, false, b.v,
                                                 (short)0, c, false, false);
}

// ---------------- utility kernels ----------------
__global__ void k_zero_counts(int* counts) { counts[threadIdx.x] = 0; }

__global__ void k_cvt_x(const float* __restrict__ x, uint16_t* __restrict__ xb) {
  const int i = (blockIdx.x * 256 + threadIdx.x) * 4;
  float4 v = *(const float4*)(x + i);
  ushort4 o; o.x = f2bf(v.x); o.y = f2bf(v.y); o.z = f2bf(v.z); o.w = f2bf(v.w);
  *(ushort4*)(xb + i) = o;
}

__global__ void k_scan(const int* __restrict__ counts, int* __restrict__ basep) {
  const int l = threadIdx.x;
  int s = 0;
  for (int j = 0; j < l; ++j) s += counts[j];
  basep[l] = s;
}

// ---------------- router: one token per wave, lane == expert ----------------
__global__ __launch_bounds__(128) void k_router(
    const float* __restrict__ x, const float* __restrict__ gw,
    const float* __restrict__ gbias, int* __restrict__ counts,
    int* __restrict__ toks, float* __restrict__ wgts) {
  __shared__ float xs[4][H_DIM];
  const int w = threadIdx.x >> 5, lane = threadIdx.x & 31;
  const int t = blockIdx.x * 4 + w;
  for (int h = lane * 4; h < H_DIM; h += 128)
    *(float4*)&xs[w][h] = *(const float4*)&x[(size_t)t * H_DIM + h];
  __syncthreads();

  float acc = 0.f;
  const float4* g4 = (const float4*)(gw + (size_t)lane * H_DIM);
  const float4* x4 = (const float4*)(xs[w]);
  for (int h = 0; h < H_DIM / 4; ++h) {
    float4 a = x4[h], b = g4[h];
    acc += a.x * b.x + a.y * b.y + a.z * b.z + a.w * b.w;
  }
  const float score = __builtin_amdgcn_rcpf(1.f + __expf(-acc));
  const float sc = score + gbias[lane];

  // group score = sum of top-2 of the 4 experts in this lane's group
  const int g = lane >> 2;
  float v0 = __shfl(sc, g * 4 + 0), v1 = __shfl(sc, g * 4 + 1);
  float v2 = __shfl(sc, g * 4 + 2), v3 = __shfl(sc, g * 4 + 3);
  float m1 = fmaxf(fmaxf(v0, v1), fmaxf(v2, v3));
  float m2;
  if      (m1 == v0) m2 = fmaxf(fmaxf(v1, v2), v3);
  else if (m1 == v1) m2 = fmaxf(fmaxf(v0, v2), v3);
  else if (m1 == v2) m2 = fmaxf(fmaxf(v0, v1), v3);
  else               m2 = fmaxf(fmaxf(v0, v1), v2);
  const float gsc = m1 + m2;

  int grank = 0;
  for (int j = 0; j < G_NUM; ++j) {
    float gj = __shfl(gsc, j * 4);
    grank += (gj > gsc) || (gj == gsc && j < g);
  }
  const float masked = (grank < TOPKG) ? sc : -3.0e38f;

  int rank = 0;
  for (int j = 0; j < 32; ++j) {
    float mj = __shfl(masked, j);
    rank += (mj > masked) || (mj == masked && j < lane);
  }
  const bool sel = (rank < K_TOP) && (masked > -1.0e38f);

  float wsum = sel ? score : 0.f;
  for (int off = 16; off; off >>= 1) wsum += __shfl_xor(wsum, off);

  if (sel) {
    const float wgt = score / wsum;
    const int pos = atomicAdd(&counts[lane], 1);
    toks[lane * T_TOK + pos] = t;
    wgts[lane * T_TOK + pos] = wgt;
  }
}

// ---------------- shared expert gate_up (pipelined, double-buffered) ----------------
__global__ __launch_bounds__(256) void k_shared_gateup(
    const uint16_t* __restrict__ xb, const float* __restrict__ sw1,
    uint16_t* __restrict__ hs) {
  __shared__ uint16_t lA[2][64 * 32];
  __shared__ uint16_t lBg[2][32 * LBS];
  __shared__ uint16_t lBu[2][32 * LBS];
  const int tid = threadIdx.x, lane = tid & 31, w = tid >> 5;
  const int m0 = blockIdx.x * 64, n0 = blockIdx.y * 32;
  const int mo = (w >> 1) * 16, no = (w & 1) * 16;
  const int ar = tid >> 2, ac = (tid & 3) * 8;
  const uint32_t aoff0 = lds_addr(&lA[0][ar * 32 + ac]);
  const uint32_t aoff1 = lds_addr(&lA[1][ar * 32 + ac]);
  const uint16_t* asrc = xb + (size_t)(m0 + ar) * H_DIM + ac;
  const int bc = tid & 31, br = (tid >> 5) * 4;
  const int bst = bc * LBS + br;
  const size_t ldb = 2 * IS_DIM;

  // prologue: fill stage 0
  async_b128(aoff0, asrc);
  {
    const float* gp = sw1 + (size_t)br * ldb + n0 + bc;
    Pack4 pg, pu;
    pg.h[0] = f2bf(gp[0]);          pg.h[1] = f2bf(gp[ldb]);
    pg.h[2] = f2bf(gp[2 * ldb]);    pg.h[3] = f2bf(gp[3 * ldb]);
    pu.h[0] = f2bf(gp[IS_DIM]);           pu.h[1] = f2bf(gp[IS_DIM + ldb]);
    pu.h[2] = f2bf(gp[IS_DIM + 2 * ldb]); pu.h[3] = f2bf(gp[IS_DIM + 3 * ldb]);
    *(unsigned long long*)&lBg[0][bst] = pg.q;
    *(unsigned long long*)&lBu[0][bst] = pu.q;
  }
  wait_asynccnt0();
  __syncthreads();

  v8f accg = {}, accu = {};
  int p = 0;
  for (int kk = 0; kk < H_DIM; kk += 32) {
    const int kn = kk + 32;
    const bool more = kn < H_DIM;
    float g0, g1, g2, g3, u0, u1, u2, u3;
    if (more) {  // issue next-stage loads (overlap with WMMA below)
      async_b128(p ? aoff0 : aoff1, asrc + kn);
      const float* gp = sw1 + (size_t)(kn + br) * ldb + n0 + bc;
      g0 = gp[0]; g1 = gp[ldb]; g2 = gp[2 * ldb]; g3 = gp[3 * ldb];
      u0 = gp[IS_DIM]; u1 = gp[IS_DIM + ldb];
      u2 = gp[IS_DIM + 2 * ldb]; u3 = gp[IS_DIM + 3 * ldb];
      if (kn + 32 < H_DIM) __builtin_prefetch(gp + 32 * ldb, 0, 1);
    }
    ABFrag a, bg, bu;
    load_afrag(lA[p], mo, lane, a);
    load_bfragT(lBg[p], no, lane, bg);
    load_bfragT(lBu[p], no, lane, bu);
    accg = wmma_bf16(a, bg, accg);
    accu = wmma_bf16(a, bu, accu);
    if (more) {
      Pack4 pg, pu;
      pg.h[0] = f2bf(g0); pg.h[1] = f2bf(g1); pg.h[2] = f2bf(g2); pg.h[3] = f2bf(g3);
      pu.h[0] = f2bf(u0); pu.h[1] = f2bf(u1); pu.h[2] = f2bf(u2); pu.h[3] = f2bf(u3);
      *(unsigned long long*)&lBg[p ^ 1][bst] = pg.q;
      *(unsigned long long*)&lBu[p ^ 1][bst] = pu.q;
      wait_asynccnt0();
    }
    __syncthreads();
    p ^= 1;
  }
  const int row0 = m0 + mo + ((lane >> 4) << 3);
  const int col = n0 + no + (lane & 15);
#pragma unroll
  for (int j = 0; j < 8; ++j)
    hs[(size_t)(row0 + j) * IS_DIM + col] = f2bf(silu_mul(accg[j], accu[j]));
}

// ---------------- shared expert down (pipelined) ----------------
__global__ __launch_bounds__(256) void k_shared_down(
    const uint16_t* __restrict__ hs, const float* __restrict__ sw2,
    float* __restrict__ out) {
  __shared__ uint16_t lA[2][64 * 32];
  __shared__ uint16_t lB[2][64 * LBS];
  const int tid = threadIdx.x, lane = tid & 31, w = tid >> 5;
  const int m0 = blockIdx.x * 64, n0 = blockIdx.y * 64;
  const int mo0 = (w >> 2) * 16, mo1 = mo0 + 32, no = (w & 3) * 16;
  const int ar = tid >> 2, ac = (tid & 3) * 8;
  const uint32_t aoff0 = lds_addr(&lA[0][ar * 32 + ac]);
  const uint32_t aoff1 = lds_addr(&lA[1][ar * 32 + ac]);
  const uint16_t* asrc = hs + (size_t)(m0 + ar) * IS_DIM + ac;
  const int bc = tid & 63, br = (tid >> 6) * 8;
  const int bst = bc * LBS + br;

  async_b128(aoff0, asrc);
  {
    const float* bp = sw2 + (size_t)br * H_DIM + n0 + bc;
    Pack8 pk;
#pragma unroll
    for (int i = 0; i < 8; ++i) pk.h[i] = f2bf(bp[(size_t)i * H_DIM]);
    *(uint4*)&lB[0][bst] = pk.q;
  }
  wait_asynccnt0();
  __syncthreads();

  v8f acc0 = {}, acc1 = {};
  int p = 0;
  for (int kk = 0; kk < IS_DIM; kk += 32) {
    const int kn = kk + 32;
    const bool more = kn < IS_DIM;
    float f[8];
    if (more) {
      async_b128(p ? aoff0 : aoff1, asrc + kn);
      const float* bp = sw2 + (size_t)(kn + br) * H_DIM + n0 + bc;
#pragma unroll
      for (int i = 0; i < 8; ++i) f[i] = bp[(size_t)i * H_DIM];
      if (kn + 32 < IS_DIM) __builtin_prefetch(bp + (size_t)32 * H_DIM, 0, 1);
    }
    ABFrag a0, a1, b;
    load_afrag(lA[p], mo0, lane, a0);
    load_afrag(lA[p], mo1, lane, a1);
    load_bfragT(lB[p], no, lane, b);
    acc0 = wmma_bf16(a0, b, acc0);
    acc1 = wmma_bf16(a1, b, acc1);
    if (more) {
      Pack8 pk;
#pragma unroll
      for (int i = 0; i < 8; ++i) pk.h[i] = f2bf(f[i]);
      *(uint4*)&lB[p ^ 1][bst] = pk.q;
      wait_asynccnt0();
    }
    __syncthreads();
    p ^= 1;
  }
  const int col = n0 + no + (lane & 15);
  const int r0 = m0 + mo0 + ((lane >> 4) << 3);
  const int r1 = m0 + mo1 + ((lane >> 4) << 3);
#pragma unroll
  for (int j = 0; j < 8; ++j) {
    out[(size_t)(r0 + j) * H_DIM + col] = acc0[j];
    out[(size_t)(r1 + j) * H_DIM + col] = acc1[j];
  }
}

// ---------------- routed gate_up (grouped GEMM, gathered rows, pipelined) ----------------
__global__ __launch_bounds__(256) void k_routed_gateup(
    const uint16_t* __restrict__ xb, const float* __restrict__ w1,
    const int* __restrict__ counts, const int* __restrict__ basep,
    const int* __restrict__ toks, uint16_t* __restrict__ hbuf) {
  const int e = blockIdx.x >> 5;
  const int mblk = blockIdx.x & 31;
  const int cnt = counts[e];
  if (mblk * 64 >= cnt) return;                  // block-uniform early exit
  const float* W = w1 + (size_t)e * H_DIM * (2 * I_DIM);
  const int rbase = basep[e];

  __shared__ uint16_t lA[2][64 * 32];
  __shared__ uint16_t lBg[2][32 * LBS];
  __shared__ uint16_t lBu[2][32 * LBS];
  __shared__ int ltok[64];
  const int tid = threadIdx.x, lane = tid & 31, w = tid >> 5;
  const int n0 = blockIdx.y * 32;
  const int mo = (w >> 1) * 16, no = (w & 1) * 16;

  if (tid < 64) {
    int r = mblk * 64 + tid;
    ltok[tid] = (r < cnt) ? toks[e * T_TOK + r] : 0;  // clamp; epilogue masks
  }
  __syncthreads();
  const int ar = tid >> 2, ac = (tid & 3) * 8;
  const uint32_t aoff0 = lds_addr(&lA[0][ar * 32 + ac]);
  const uint32_t aoff1 = lds_addr(&lA[1][ar * 32 + ac]);
  const uint16_t* asrc = xb + (size_t)ltok[ar] * H_DIM + ac;
  const int bc = tid & 31, br = (tid >> 5) * 4;
  const int bst = bc * LBS + br;
  const size_t ldb = 2 * I_DIM;

  async_b128(aoff0, asrc);
  {
    const float* gp = W + (size_t)br * ldb + n0 + bc;
    Pack4 pg, pu;
    pg.h[0] = f2bf(gp[0]);       pg.h[1] = f2bf(gp[ldb]);
    pg.h[2] = f2bf(gp[2 * ldb]); pg.h[3] = f2bf(gp[3 * ldb]);
    pu.h[0] = f2bf(gp[I_DIM]);           pu.h[1] = f2bf(gp[I_DIM + ldb]);
    pu.h[2] = f2bf(gp[I_DIM + 2 * ldb]); pu.h[3] = f2bf(gp[I_DIM + 3 * ldb]);
    *(unsigned long long*)&lBg[0][bst] = pg.q;
    *(unsigned long long*)&lBu[0][bst] = pu.q;
  }
  wait_asynccnt0();
  __syncthreads();

  v8f accg = {}, accu = {};
  int p = 0;
  for (int kk = 0; kk < H_DIM; kk += 32) {
    const int kn = kk + 32;
    const bool more = kn < H_DIM;
    float g0, g1, g2, g3, u0, u1, u2, u3;
    if (more) {
      async_b128(p ? aoff0 : aoff1, asrc + kn);
      const float* gp = W + (size_t)(kn + br) * ldb + n0 + bc;
      g0 = gp[0]; g1 = gp[ldb]; g2 = gp[2 * ldb]; g3 = gp[3 * ldb];
      u0 = gp[I_DIM]; u1 = gp[I_DIM + ldb];
      u2 = gp[I_DIM + 2 * ldb]; u3 = gp[I_DIM + 3 * ldb];
      if (kn + 32 < H_DIM) __builtin_prefetch(gp + 32 * ldb, 0, 1);
    }
    ABFrag a, bg, bu;
    load_afrag(lA[p], mo, lane, a);
    load_bfragT(lBg[p], no, lane, bg);
    load_bfragT(lBu[p], no, lane, bu);
    accg = wmma_bf16(a, bg, accg);
    accu = wmma_bf16(a, bu, accu);
    if (more) {
      Pack4 pg, pu;
      pg.h[0] = f2bf(g0); pg.h[1] = f2bf(g1); pg.h[2] = f2bf(g2); pg.h[3] = f2bf(g3);
      pu.h[0] = f2bf(u0); pu.h[1] = f2bf(u1); pu.h[2] = f2bf(u2); pu.h[3] = f2bf(u3);
      *(unsigned long long*)&lBg[p ^ 1][bst] = pg.q;
      *(unsigned long long*)&lBu[p ^ 1][bst] = pu.q;
      wait_asynccnt0();
    }
    __syncthreads();
    p ^= 1;
  }
  const int rloc = mblk * 64 + mo + ((lane >> 4) << 3);
  const int col = n0 + no + (lane & 15);
#pragma unroll
  for (int j = 0; j < 8; ++j) {
    int r = rloc + j;
    if (r < cnt)
      hbuf[(size_t)(rbase + r) * I_DIM + col] = f2bf(silu_mul(accg[j], accu[j]));
  }
}

// ---------------- routed down + weighted scatter-add (pipelined) ----------------
__global__ __launch_bounds__(256) void k_routed_down(
    const uint16_t* __restrict__ hbuf, const float* __restrict__ w2,
    const int* __restrict__ counts, const int* __restrict__ basep,
    const int* __restrict__ toks, const float* __restrict__ wgts,
    float* __restrict__ out) {
  const int e = blockIdx.x >> 5;
  const int mblk = blockIdx.x & 31;
  const int cnt = counts[e];
  if (mblk * 64 >= cnt) return;
  const float* W = w2 + (size_t)e * I_DIM * H_DIM;
  const int rbase = basep[e];

  __shared__ uint16_t lA[2][64 * 32];
  __shared__ uint16_t lB[2][64 * LBS];
  const int tid = threadIdx.x, lane = tid & 31, w = tid >> 5;
  const int n0 = blockIdx.y * 64;
  const int mo0 = (w >> 2) * 16, mo1 = mo0 + 32, no = (w & 3) * 16;

  const int ar = tid >> 2, ac = (tid & 3) * 8;
  int arow = mblk * 64 + ar; if (arow >= cnt) arow = cnt - 1;   // clamp; epilogue masks
  const uint32_t aoff0 = lds_addr(&lA[0][ar * 32 + ac]);
  const uint32_t aoff1 = lds_addr(&lA[1][ar * 32 + ac]);
  const uint16_t* asrc = hbuf + (size_t)(rbase + arow) * I_DIM + ac;
  const int bc = tid & 63, br = (tid >> 6) * 8;
  const int bst = bc * LBS + br;

  async_b128(aoff0, asrc);
  {
    const float* bp = W + (size_t)br * H_DIM + n0 + bc;
    Pack8 pk;
#pragma unroll
    for (int i = 0; i < 8; ++i) pk.h[i] = f2bf(bp[(size_t)i * H_DIM]);
    *(uint4*)&lB[0][bst] = pk.q;
  }
  wait_asynccnt0();
  __syncthreads();

  v8f acc0 = {}, acc1 = {};
  int p = 0;
  for (int kk = 0; kk < I_DIM; kk += 32) {
    const int kn = kk + 32;
    const bool more = kn < I_DIM;
    float f[8];
    if (more) {
      async_b128(p ? aoff0 : aoff1, asrc + kn);
      const float* bp = W + (size_t)(kn + br) * H_DIM + n0 + bc;
#pragma unroll
      for (int i = 0; i < 8; ++i) f[i] = bp[(size_t)i * H_DIM];
      if (kn + 32 < I_DIM) __builtin_prefetch(bp + (size_t)32 * H_DIM, 0, 1);
    }
    ABFrag a0, a1, b;
    load_afrag(lA[p], mo0, lane, a0);
    load_afrag(lA[p], mo1, lane, a1);
    load_bfragT(lB[p], no, lane, b);
    acc0 = wmma_bf16(a0, b, acc0);
    acc1 = wmma_bf16(a1, b, acc1);
    if (more) {
      Pack8 pk;
#pragma unroll
      for (int i = 0; i < 8; ++i) pk.h[i] = f2bf(f[i]);
      *(uint4*)&lB[p ^ 1][bst] = pk.q;
      wait_asynccnt0();
    }
    __syncthreads();
    p ^= 1;
  }
  const int col = n0 + no + (lane & 15);
  const int r0 = mblk * 64 + mo0 + ((lane >> 4) << 3);
  const int r1 = mblk * 64 + mo1 + ((lane >> 4) << 3);
#pragma unroll
  for (int j = 0; j < 8; ++j) {
    int ra = r0 + j;
    if (ra < cnt) {
      int t = toks[e * T_TOK + ra];
      float s = SCALE_F * wgts[e * T_TOK + ra];
      atomicAdd(&out[(size_t)t * H_DIM + col], acc0[j] * s);
    }
    int rb = r1 + j;
    if (rb < cnt) {
      int t = toks[e * T_TOK + rb];
      float s = SCALE_F * wgts[e * T_TOK + rb];
      atomicAdd(&out[(size_t)t * H_DIM + col], acc1[j] * s);
    }
  }
}

// ---------------- host-side launch ----------------
extern "C" void kernel_launch(void* const* d_in, const int* in_sizes, int n_in,
                              void* d_out, int out_size, void* d_ws, size_t ws_size,
                              hipStream_t stream) {
  const float* x     = (const float*)d_in[0];
  const float* gw    = (const float*)d_in[1];
  const float* gbias = (const float*)d_in[2];
  const float* w1    = (const float*)d_in[3];
  const float* w2    = (const float*)d_in[4];
  const float* sw1   = (const float*)d_in[5];
  const float* sw2   = (const float*)d_in[6];
  float* out = (float*)d_out;

  // workspace layout (bytes)
  char* ws = (char*)d_ws;
  uint16_t* xb    = (uint16_t*)(ws);                               // T*H bf16 = 8 MB
  size_t off = (size_t)T_TOK * H_DIM * 2;
  int* counts     = (int*)(ws + off);          off += 256;
  int* basep      = (int*)(ws + off);          off += 256;
  int* toks       = (int*)(ws + off);          off += (size_t)E_NUM * T_TOK * 4;
  float* wgts     = (float*)(ws + off);        off += (size_t)E_NUM * T_TOK * 4;
  uint16_t* hbuf  = (uint16_t*)(ws + off);     off += (size_t)T_TOK * K_TOP * I_DIM * 2;
  uint16_t* hsbuf = (uint16_t*)(ws + off);     // T*IS bf16 = 11.5 MB

  k_zero_counts<<<1, E_NUM, 0, stream>>>(counts);
  k_cvt_x<<<(T_TOK * H_DIM) / (256 * 4), 256, 0, stream>>>(x, xb);
  k_router<<<T_TOK / 4, 128, 0, stream>>>(x, gw, gbias, counts, toks, wgts);
  k_scan<<<1, E_NUM, 0, stream>>>(counts, basep);
  k_shared_gateup<<<dim3(T_TOK / 64, IS_DIM / 32), 256, 0, stream>>>(xb, sw1, hsbuf);
  k_shared_down<<<dim3(T_TOK / 64, H_DIM / 64), 256, 0, stream>>>(hsbuf, sw2, out);
  k_routed_gateup<<<dim3(E_NUM * 32, I_DIM / 32), 256, 0, stream>>>(
      xb, w1, counts, basep, toks, hbuf);
  k_routed_down<<<dim3(E_NUM * 32, H_DIM / 64), 256, 0, stream>>>(
      hbuf, w2, counts, basep, toks, wgts, out);
}
